// NSAGE_6098853560421
// MI455X (gfx1250) — compile-verified, matrix operand order checked by reading, not run on hardware
//
#include <hip/hip_runtime.h>

#define NN 10000
#define NE 80000
#define DIN 1024
#define DH  4096
#define DOUT 256

typedef __attribute__((ext_vector_type(16))) __bf16 v16bf;
typedef __attribute__((ext_vector_type(8)))  float  v8f;
typedef __attribute__((ext_vector_type(4)))  float  v4f;

union Frag { v16bf v; v4f f4[2]; };

// ---------------- utility kernels ----------------

__global__ void k_zero_b32(unsigned* __restrict__ p, long n) {
    long i = (long)blockIdx.x * blockDim.x + threadIdx.x;
    long s = (long)gridDim.x * blockDim.x;
    for (; i < n; i += s) p[i] = 0u;
}

__global__ void k_cast_bf16(const float* __restrict__ in, __bf16* __restrict__ out, long n) {
    long i = (long)blockIdx.x * blockDim.x + threadIdx.x;
    long s = (long)gridDim.x * blockDim.x;
    for (; i < n; i += s) out[i] = (__bf16)in[i];
}

// ---------------- CSR build: degree count, exclusive scan, bucket fill ----------------

__global__ void k_count_deg(const int* __restrict__ dst, int* __restrict__ deg, int E) {
    int e = blockIdx.x * blockDim.x + threadIdx.x;
    if (e < E) atomicAdd(&deg[dst[e]], 1);
}

// single-block (1024-thread) exclusive scan over NN degrees
__global__ __launch_bounds__(1024) void k_scan(const int* __restrict__ deg,
                                               int* __restrict__ rowstart,
                                               int* __restrict__ cursor) {
    __shared__ int buf[1024];
    __shared__ int carry;
    if (threadIdx.x == 0) carry = 0;
    __syncthreads();
    for (int base = 0; base < NN; base += 1024) {
        int i = base + (int)threadIdx.x;
        int v = (i < NN) ? deg[i] : 0;
        buf[threadIdx.x] = v;
        __syncthreads();
        for (int off = 1; off < 1024; off <<= 1) {       // Hillis-Steele inclusive scan
            int t = (threadIdx.x >= (unsigned)off) ? buf[threadIdx.x - off] : 0;
            __syncthreads();
            buf[threadIdx.x] += t;
            __syncthreads();
        }
        int excl = buf[threadIdx.x] - v + carry;
        if (i < NN) { rowstart[i] = excl; cursor[i] = excl; }
        __syncthreads();
        if (threadIdx.x == 1023) carry += buf[1023];
        __syncthreads();
    }
    if (threadIdx.x == 0) rowstart[NN] = carry;          // == NE
}

__global__ void k_fill(const int* __restrict__ src, const int* __restrict__ dst,
                       int* __restrict__ cursor, int* __restrict__ elist, int E) {
    int e = blockIdx.x * blockDim.x + threadIdx.x;
    if (e < E) {
        int pos = atomicAdd(&cursor[dst[e]], 1);
        elist[pos] = src[e];
    }
}

// ---------------- fused gather + mean + bf16 cast (no FP atomics) ----------------
// One block per destination node. Neighbor ids staged in LDS; neighbor rows
// streamed with coalesced b64 loads; fp32 accumulation in registers.
// D/1024 must be an integer (DIN: 1, DH: 4).

__global__ __launch_bounds__(256) void k_gather_mean(
    const __bf16* __restrict__ X, const int* __restrict__ rowstart,
    const int* __restrict__ elist, __bf16* __restrict__ out, int D)
{
    __shared__ int nb[256];
    int n   = blockIdx.x;
    int beg = rowstart[n];
    int end = rowstart[n + 1];
    int deg = end - beg;
    float inv = (deg > 0) ? 1.0f / (float)deg : 0.0f;    // isolated node -> mean 0

    int nIter = D >> 10;                                  // D / (256 threads * 4 bf16)
    float acc[4][4];
#pragma unroll
    for (int it = 0; it < 4; ++it)
#pragma unroll
        for (int c = 0; c < 4; ++c) acc[it][c] = 0.0f;

    for (int chunk = beg; chunk < end; chunk += 256) {
        int cnt = end - chunk;
        if (cnt > 256) cnt = 256;
        if ((int)threadIdx.x < cnt) nb[threadIdx.x] = elist[chunk + threadIdx.x];
        __syncthreads();
        for (int j = 0; j < cnt; ++j) {
            const uint2* row = (const uint2*)(X + (long)nb[j] * D);
            for (int it = 0; it < nIter; ++it) {
                uint2 p = row[threadIdx.x + (it << 8)];   // 4 bf16, coalesced b64
                acc[it][0] += __builtin_bit_cast(float, p.x << 16);
                acc[it][1] += __builtin_bit_cast(float, p.x & 0xffff0000u);
                acc[it][2] += __builtin_bit_cast(float, p.y << 16);
                acc[it][3] += __builtin_bit_cast(float, p.y & 0xffff0000u);
            }
        }
        __syncthreads();
    }

    uint2* orow = (uint2*)(out + (long)n * D);
    for (int it = 0; it < nIter; ++it) {
        unsigned short s0 = __builtin_bit_cast(unsigned short, (__bf16)(acc[it][0] * inv));
        unsigned short s1 = __builtin_bit_cast(unsigned short, (__bf16)(acc[it][1] * inv));
        unsigned short s2 = __builtin_bit_cast(unsigned short, (__bf16)(acc[it][2] * inv));
        unsigned short s3 = __builtin_bit_cast(unsigned short, (__bf16)(acc[it][3] * inv));
        uint2 o;
        o.x = (unsigned)s0 | ((unsigned)s1 << 16);
        o.y = (unsigned)s2 | ((unsigned)s3 << 16);
        orow[threadIdx.x + (it << 8)] = o;
    }
}

// ---------------- dual-operand WMMA GEMM, 32x32 register tile ----------------
// out[M,N] = A1[M,K] @ B1[N,K]^T + A2[M,K] @ B2[N,K]^T + bias[N]  (opt relu)
// One wave computes a 32x32 output tile: 2 M-tiles x 2 N-tiles, 4 WMMA per
// K-step with both A and B fragments reused twice (4 wmma : 8 b128 loads).
// Fragment packing per CDNA5 ISA 7.12.2 (16-bit A 16x32 / B 32x16):
//   r16 = lane&15 -> row(A)/col(B); kh8 = (lane>>4)*8.
//   v16bf elems 0..7  = K(k0 + kh8 + 0..7), elems 8..15 = K(k0 + 16 + kh8 + 0..7)

#define WMMA_BF16(A, B, C) \
    __builtin_amdgcn_wmma_f32_16x16x32_bf16(false, (A).v, false, (B).v, (short)0, (C), false, false)

__global__ __launch_bounds__(256) void k_gemm_dual(
    const __bf16* __restrict__ A1, const __bf16* __restrict__ B1,
    const __bf16* __restrict__ A2, const __bf16* __restrict__ B2,
    const float*  __restrict__ bias,
    float* __restrict__ outF, __bf16* __restrict__ outB,
    int M, int N, int K, int relu)
{
    int w      = (blockIdx.x * blockDim.x + threadIdx.x) >> 5;
    int lane   = threadIdx.x & 31;
    int mTiles = M >> 4;
    int mPairs = (mTiles + 1) >> 1;
    int nPairs = N >> 5;
    int total  = mPairs * nPairs;
    if (w >= total) return;

    int mt0  = (w / nPairs) * 2;
    int mt1  = mt0 + 1;
    int m1ok = (mt1 < mTiles);        // wave-uniform; tail pair duplicates mt0
    if (!m1ok) mt1 = mt0;
    int nt0  = (w % nPairs) * 2;

    int r16 = lane & 15;
    int kh8 = (lane >> 4) * 8;

    const __bf16* pa0 = A1 + (long)(mt0 * 16 + r16) * K + kh8;
    const __bf16* pa1 = A1 + (long)(mt1 * 16 + r16) * K + kh8;
    const __bf16* pb0 = B1 + (long)(nt0 * 16 + r16) * K + kh8;
    const __bf16* pb1 = B1 + (long)((nt0 + 1) * 16 + r16) * K + kh8;
    const __bf16* qa0 = A2 + (long)(mt0 * 16 + r16) * K + kh8;
    const __bf16* qa1 = A2 + (long)(mt1 * 16 + r16) * K + kh8;
    const __bf16* qb0 = B2 + (long)(nt0 * 16 + r16) * K + kh8;
    const __bf16* qb1 = B2 + (long)((nt0 + 1) * 16 + r16) * K + kh8;

    v8f c00 = {}, c01 = {}, c10 = {}, c11 = {};

    for (int k0 = 0; k0 < K; k0 += 32) {
        Frag a0, a1, b0, b1;
        a0.f4[0] = *(const v4f*)(pa0 + k0); a0.f4[1] = *(const v4f*)(pa0 + k0 + 16);
        a1.f4[0] = *(const v4f*)(pa1 + k0); a1.f4[1] = *(const v4f*)(pa1 + k0 + 16);
        b0.f4[0] = *(const v4f*)(pb0 + k0); b0.f4[1] = *(const v4f*)(pb0 + k0 + 16);
        b1.f4[0] = *(const v4f*)(pb1 + k0); b1.f4[1] = *(const v4f*)(pb1 + k0 + 16);
        __builtin_prefetch(pa0 + k0 + 64, 0, 1);   // global_prefetch_b8 next K block
        __builtin_prefetch(pa1 + k0 + 64, 0, 1);
        c00 = WMMA_BF16(a0, b0, c00);
        c01 = WMMA_BF16(a0, b1, c01);
        c10 = WMMA_BF16(a1, b0, c10);
        c11 = WMMA_BF16(a1, b1, c11);
    }
    for (int k0 = 0; k0 < K; k0 += 32) {
        Frag a0, a1, b0, b1;
        a0.f4[0] = *(const v4f*)(qa0 + k0); a0.f4[1] = *(const v4f*)(qa0 + k0 + 16);
        a1.f4[0] = *(const v4f*)(qa1 + k0); a1.f4[1] = *(const v4f*)(qa1 + k0 + 16);
        b0.f4[0] = *(const v4f*)(qb0 + k0); b0.f4[1] = *(const v4f*)(qb0 + k0 + 16);
        b1.f4[0] = *(const v4f*)(qb1 + k0); b1.f4[1] = *(const v4f*)(qb1 + k0 + 16);
        __builtin_prefetch(qa0 + k0 + 64, 0, 1);
        __builtin_prefetch(qa1 + k0 + 64, 0, 1);
        c00 = WMMA_BF16(a0, b0, c00);
        c01 = WMMA_BF16(a0, b1, c01);
        c10 = WMMA_BF16(a1, b0, c10);
        c11 = WMMA_BF16(a1, b1, c11);
    }

    // C/D layout: VGPR r, lanes 0-15 -> M = r, lanes 16-31 -> M = r + 8; N = r16
    int n0 = nt0 * 16 + r16;
    int n1 = n0 + 16;
    float bs0 = bias[n0];
    float bs1 = bias[n1];
    int khalf = lane >> 4;
#pragma unroll
    for (int r = 0; r < 8; ++r) {
        int m0 = mt0 * 16 + khalf * 8 + r;
        int m1 = mt1 * 16 + khalf * 8 + r;
        float v00 = c00[r] + bs0, v01 = c01[r] + bs1;
        float v10 = c10[r] + bs0, v11 = c11[r] + bs1;
        if (relu) {
            v00 = v00 > 0.f ? v00 : 0.f;  v01 = v01 > 0.f ? v01 : 0.f;
            v10 = v10 > 0.f ? v10 : 0.f;  v11 = v11 > 0.f ? v11 : 0.f;
        }
        if (outB) {
            outB[(long)m0 * N + n0] = (__bf16)v00;
            outB[(long)m0 * N + n1] = (__bf16)v01;
            if (m1ok) {
                outB[(long)m1 * N + n0] = (__bf16)v10;
                outB[(long)m1 * N + n1] = (__bf16)v11;
            }
        } else {
            outF[(long)m0 * N + n0] = v00;
            outF[(long)m0 * N + n1] = v01;
            if (m1ok) {
                outF[(long)m1 * N + n0] = v10;
                outF[(long)m1 * N + n1] = v11;
            }
        }
    }
}

// ---------------- in-place log-softmax, one wave32 per 256-wide row ----------------
__global__ void k_logsoftmax(float* __restrict__ out) {
    int w    = (blockIdx.x * blockDim.x + threadIdx.x) >> 5;
    int lane = threadIdx.x & 31;
    if (w >= NN) return;
    float* row = out + (long)w * DOUT;
    float v[8];
    float mx = -3.4e38f;
#pragma unroll
    for (int j = 0; j < 8; ++j) { v[j] = row[lane + 32 * j]; mx = fmaxf(mx, v[j]); }
    for (int off = 16; off > 0; off >>= 1) mx = fmaxf(mx, __shfl_xor(mx, off, 32));
    float s = 0.f;
#pragma unroll
    for (int j = 0; j < 8; ++j) s += __expf(v[j] - mx);
    for (int off = 16; off > 0; off >>= 1) s += __shfl_xor(s, off, 32);
    float ls = __logf(s);
#pragma unroll
    for (int j = 0; j < 8; ++j) row[lane + 32 * j] = v[j] - mx - ls;
}

// ---------------- host orchestration ----------------

extern "C" void kernel_launch(void* const* d_in, const int* in_sizes, int n_in,
                              void* d_out, int out_size, void* d_ws, size_t ws_size,
                              hipStream_t stream) {
    const float* x   = (const float*)d_in[0];
    const int*   ei  = (const int*)  d_in[1];
    const float* W1l = (const float*)d_in[2];
    const float* b1l = (const float*)d_in[3];
    const float* W1r = (const float*)d_in[4];
    const float* W2l = (const float*)d_in[5];
    const float* b2l = (const float*)d_in[6];
    const float* W2r = (const float*)d_in[7];
    float* out = (float*)d_out;
    const int* src = ei;
    const int* dst = ei + NE;

    char* ws = (char*)d_ws;
    size_t cur = 0;
    auto carve = [&](size_t bytes) -> void* {
        void* p = ws + cur;
        cur = (cur + bytes + 255) & ~(size_t)255;
        return p;
    };

    __bf16* xbf      = (__bf16*)carve((size_t)NN * DIN * 2);
    __bf16* meanbf   = (__bf16*)carve((size_t)NN * DIN * 2);
    __bf16* w1lb     = (__bf16*)carve((size_t)DH * DIN * 2);
    __bf16* w1rb     = (__bf16*)carve((size_t)DH * DIN * 2);
    __bf16* w2lb     = (__bf16*)carve((size_t)DOUT * DH * 2);
    __bf16* w2rb     = (__bf16*)carve((size_t)DOUT * DH * 2);
    __bf16* hbf      = (__bf16*)carve((size_t)NN * DH * 2);
    __bf16* meanhbf  = (__bf16*)carve((size_t)NN * DH * 2);
    int*    deg      = (int*)   carve((size_t)NN * 4);
    int*    rowstart = (int*)   carve((size_t)(NN + 1) * 4);
    int*    cursor   = (int*)   carve((size_t)NN * 4);
    int*    elist    = (int*)   carve((size_t)NE * 4);

    // casts to bf16
    k_cast_bf16<<<4096, 256, 0, stream>>>(x,   xbf,  (long)NN * DIN);
    k_cast_bf16<<<4096, 256, 0, stream>>>(W1l, w1lb, (long)DH * DIN);
    k_cast_bf16<<<4096, 256, 0, stream>>>(W1r, w1rb, (long)DH * DIN);
    k_cast_bf16<<<2048, 256, 0, stream>>>(W2l, w2lb, (long)DOUT * DH);
    k_cast_bf16<<<2048, 256, 0, stream>>>(W2r, w2rb, (long)DOUT * DH);

    // CSR build (must be rebuilt every call: deg is accumulated with atomics)
    k_zero_b32<<<64, 256, 0, stream>>>((unsigned*)deg, (long)NN);
    k_count_deg<<<(NE + 255) / 256, 256, 0, stream>>>(dst, deg, NE);
    k_scan<<<1, 1024, 0, stream>>>(deg, rowstart, cursor);
    k_fill<<<(NE + 255) / 256, 256, 0, stream>>>(src, dst, cursor, elist, NE);

    // ---- layer 1: mean of x over incoming edges (gather, no FP atomics) ----
    k_gather_mean<<<NN, 256, 0, stream>>>(xbf, rowstart, elist, meanbf, DIN);

    // ---- GEMM1: h = relu(mean @ W1_l^T + b1 + x @ W1_r^T) -> bf16 ----
    {
        int mPairs = (NN / 16 + 1) / 2;          // 313
        int total  = mPairs * (DH / 32);         // 313 * 128 = 40064 waves
        int blocks = (total + 7) / 8;            // 8 waves / block
        k_gemm_dual<<<blocks, 256, 0, stream>>>(meanbf, w1lb, xbf, w1rb, b1l,
                                                nullptr, hbf, NN, DH, DIN, 1);
    }

    // ---- layer 2: mean of h over incoming edges ----
    k_gather_mean<<<NN, 256, 0, stream>>>(hbf, rowstart, elist, meanhbf, DH);

    // ---- GEMM2: out = meanh @ W2_l^T + b2 + h @ W2_r^T -> fp32 logits ----
    {
        int mPairs = (NN / 16 + 1) / 2;          // 313
        int total  = mPairs * (DOUT / 32);       // 313 * 8 = 2504 waves
        int blocks = (total + 7) / 8;            // 313
        k_gemm_dual<<<blocks, 256, 0, stream>>>(meanhbf, w2lb, hbf, w2rb, b2l,
                                                out, nullptr, NN, DOUT, DH, 0);
    }

    // ---- log_softmax in place on d_out ----
    k_logsoftmax<<<(NN * 32 + 255) / 256, 256, 0, stream>>>(out);
}
